// MSCC_11003706212527
// MI455X (gfx1250) — matrix operand order repaired and checked by reading, not compiled, not run on hardware
//
#include <hip/hip_runtime.h>
#include <hip/hip_bf16.h>

typedef __attribute__((ext_vector_type(16))) _Float16 v16h;
typedef __attribute__((ext_vector_type(8)))  float    v8f;

#define BSZ   8
#define CCH   512
#define HW    4096
#define NUMG  64
#define CGc   8
#define NCl   19
#define OUTC  512
#define KTOT  9216              // 2C * 9
#define CHW   (CCH * HW)        // 2097152
#define C2HW  (2 * CCH * HW)    // 4194304 (cat buffer per batch)
#define LKC   (NCl * CGc)       // 152

// ---------------------------------------------------------------------------
// Precision prep: f16 copies feeding the WMMA conv.
// wgt16[o][k] = (f16) bot_w ; cat16[b][0:512][p] = (f16) x (attn fills 512:1024)
// ---------------------------------------------------------------------------
__global__ void __launch_bounds__(256)
cvt_wgt_kernel(const float* __restrict__ w, _Float16* __restrict__ w16)
{
    for (int i = blockIdx.x * 256 + threadIdx.x; i < OUTC * KTOT;
         i += gridDim.x * 256)
        w16[i] = (_Float16)w[i];
}

__global__ void __launch_bounds__(256)
cvt_x_kernel(const float* __restrict__ x, _Float16* __restrict__ cat16)
{
    const size_t i = (size_t)blockIdx.x * 256 + threadIdx.x;  // < B*C*HW
    const size_t b = i / CHW, rem = i - b * CHW;
    cat16[b * C2HW + rem] = (_Float16)x[i];
}

// ---------------------------------------------------------------------------
// Stage A1: local[b,n,k,c] = sum_p fres_map[b,n,k,p] * fg[b,n,c,p]
// Block per (b,n); [19x8] VGPR accumulator, shuffle + LDS reduction.
// ---------------------------------------------------------------------------
__global__ void __launch_bounds__(256)
stageA_local_kernel(const float* __restrict__ x, const float* __restrict__ dct,
                    const float* __restrict__ w1, const float* __restrict__ gma,
                    const float* __restrict__ bta, const float* __restrict__ w2,
                    const float* __restrict__ b2, float* __restrict__ d_local)
{
    const int n = blockIdx.x, b = blockIdx.y;
    const int tid = threadIdx.x, lane = tid & 31, wid = tid >> 5;
    __shared__ float red[8][LKC];

    float acc[NCl][CGc];
#pragma unroll
    for (int k = 0; k < NCl; ++k)
#pragma unroll
        for (int c = 0; c < CGc; ++c) acc[k][c] = 0.f;

    const float* xg = x   + (size_t)b * CHW + (size_t)n * CGc * HW;
    const float* dg = dct + (size_t)n * CGc * HW;   // basis identical across group

    for (int p = tid; p < HW; p += 256) {
        const float dv = dg[p];
        float fv[CGc];
#pragma unroll
        for (int c = 0; c < CGc; ++c) fv[c] = xg[c * HW + p] * dv;
        float h1[CGc];
#pragma unroll
        for (int d = 0; d < CGc; ++d) {
            float s = 0.f;
#pragma unroll
            for (int c = 0; c < CGc; ++c) s += w1[d * CGc + c] * fv[c];
            s = s * gma[d] + bta[d];
            h1[d] = s > 0.f ? s : 0.f;
        }
        float fm[NCl];
#pragma unroll
        for (int k = 0; k < NCl; ++k) {
            float s = b2[k];
#pragma unroll
            for (int d = 0; d < CGc; ++d) s += w2[k * CGc + d] * h1[d];
            fm[k] = s;
        }
#pragma unroll
        for (int k = 0; k < NCl; ++k)
#pragma unroll
            for (int c = 0; c < CGc; ++c) acc[k][c] += fm[k] * fv[c];
    }

#pragma unroll
    for (int k = 0; k < NCl; ++k)
#pragma unroll
        for (int c = 0; c < CGc; ++c) {
            float v = acc[k][c];
            for (int off = 16; off > 0; off >>= 1) v += __shfl_down(v, off, 32);
            if (lane == 0) red[wid][k * CGc + c] = v;
        }
    __syncthreads();
    if (tid < LKC) {
        float s = 0.f;
#pragma unroll
        for (int w = 0; w < 8; ++w) s += red[w][tid];
        d_local[((size_t)b * NUMG + n) * LKC + tid] = s;
    }
}

// ---------------------------------------------------------------------------
// Stage A2: fres_map_output[b,k,p] = sum_n fuse_w[n]*fm[b,n,k,p] + fuse_b.
// ---------------------------------------------------------------------------
__global__ void __launch_bounds__(256)
stageA_fuse_kernel(const float* __restrict__ x, const float* __restrict__ dct,
                   const float* __restrict__ w1, const float* __restrict__ gma,
                   const float* __restrict__ bta, const float* __restrict__ w2,
                   const float* __restrict__ b2, const float* __restrict__ fuse_w,
                   const float* __restrict__ fuse_b, float* __restrict__ out)
{
    const int b = blockIdx.y;
    const int p = blockIdx.x * 256 + threadIdx.x;
    const float* xb = x + (size_t)b * CHW;

    float fa[NCl];
#pragma unroll
    for (int k = 0; k < NCl; ++k) fa[k] = 0.f;

    for (int n = 0; n < NUMG; ++n) {
        const float dv = dct[(size_t)n * CGc * HW + p];
        const float fw = fuse_w[n];
        float fv[CGc];
#pragma unroll
        for (int c = 0; c < CGc; ++c) fv[c] = xb[(n * CGc + c) * HW + p] * dv;
        float h1[CGc];
#pragma unroll
        for (int d = 0; d < CGc; ++d) {
            float s = 0.f;
#pragma unroll
            for (int c = 0; c < CGc; ++c) s += w1[d * CGc + c] * fv[c];
            s = s * gma[d] + bta[d];
            h1[d] = s > 0.f ? s : 0.f;
        }
#pragma unroll
        for (int k = 0; k < NCl; ++k) {
            float s = b2[k];
#pragma unroll
            for (int d = 0; d < CGc; ++d) s += w2[k * CGc + d] * h1[d];
            fa[k] += fw * s;
        }
    }
    const float fb = fuse_b[0];
#pragma unroll
    for (int k = 0; k < NCl; ++k)
        out[((size_t)b * NCl + k) * HW + p] = fa[k] + fb;
}

// ---------------------------------------------------------------------------
// GCN: g = relu(gcn_w1 @ local + local); localF = g @ gcn_w2^T.
// ---------------------------------------------------------------------------
__global__ void __launch_bounds__(256)
gcn_kernel(const float* __restrict__ d_local, const float* __restrict__ gw1,
           const float* __restrict__ gw2, float* __restrict__ d_localF)
{
    const int b = blockIdx.x;
    __shared__ float loc[NUMG * LKC];
    const float* lb = d_local + (size_t)b * NUMG * LKC;
    for (int i = threadIdx.x; i < NUMG * LKC; i += 256) loc[i] = lb[i];
    __syncthreads();

    for (int i = threadIdx.x; i < NUMG * NCl; i += 256) {
        const int m = i / NCl, k = i % NCl;
        float gv[CGc];
#pragma unroll
        for (int c = 0; c < CGc; ++c) {
            float s = 0.f;
            for (int n2 = 0; n2 < NUMG; ++n2)
                s += gw1[m * NUMG + n2] * loc[n2 * LKC + k * CGc + c];
            s += loc[m * LKC + k * CGc + c];
            gv[c] = s > 0.f ? s : 0.f;
        }
#pragma unroll
        for (int d = 0; d < CGc; ++d) {
            float s = 0.f;
#pragma unroll
            for (int c = 0; c < CGc; ++c) s += gv[c] * gw2[d * CGc + c];
            d_localF[(size_t)b * NUMG * LKC + m * LKC + k * CGc + d] = s;
        }
    }
}

// ---------------------------------------------------------------------------
// Attention; writes its result in f16 directly into cat16 channels [512,1024).
// ---------------------------------------------------------------------------
__global__ void __launch_bounds__(256)
attn_kernel(const float* __restrict__ x, const float* __restrict__ d_localF,
            _Float16* __restrict__ cat16)
{
    const int n = blockIdx.x, b = blockIdx.y;
    __shared__ float lf[LKC];
    if (threadIdx.x < LKC)
        lf[threadIdx.x] = d_localF[((size_t)b * NUMG + n) * LKC + threadIdx.x];
    __syncthreads();

    const float* xb = x + (size_t)b * CHW + (size_t)n * CGc * HW;
    _Float16*    ab = cat16 + (size_t)b * C2HW + (size_t)(CCH + n * CGc) * HW;

    for (int p = threadIdx.x; p < HW; p += 256) {
        float xv[CGc];
#pragma unroll
        for (int c = 0; c < CGc; ++c) xv[c] = xb[c * HW + p];
        float lg[NCl];
        float mx = -3.4e38f;
#pragma unroll
        for (int k = 0; k < NCl; ++k) {
            float s = 0.f;
#pragma unroll
            for (int c = 0; c < CGc; ++c) s += xv[c] * lf[k * CGc + c];
            lg[k] = s;
            mx = s > mx ? s : mx;
        }
        float se = 0.f;
#pragma unroll
        for (int k = 0; k < NCl; ++k) { lg[k] = expf(lg[k] - mx); se += lg[k]; }
        const float inv = 1.f / se;
        float ov[CGc];
#pragma unroll
        for (int c = 0; c < CGc; ++c) ov[c] = 0.f;
#pragma unroll
        for (int k = 0; k < NCl; ++k)
#pragma unroll
            for (int c = 0; c < CGc; ++c) ov[c] += lg[k] * lf[k * CGc + c];
#pragma unroll
        for (int c = 0; c < CGc; ++c) ab[c * HW + p] = (_Float16)(ov[c] * inv);
    }
}

// ---------------------------------------------------------------------------
// Bottleneck 3x3 conv, implicit GEMM, v_wmma_f32_16x16x32_f16.
// Block tile 128(M) x 64(N); 8 waves = 4(M) x 2(N); each wave: 2 A frags x
// 2 B frags -> 4 accumulators, 4 WMMA per K-step (2x LDS-to-math vs 2 WMMA).
// Weight tile (8 KB f16) DMA'd via global_load_async_to_lds_b128 (2x16B per
// lane), overlapped with the im2col gather, fenced by s_wait_asynccnt.
// ---------------------------------------------------------------------------
__global__ void __launch_bounds__(256)
conv_wmma_kernel(const _Float16* __restrict__ cat16,
                 const _Float16* __restrict__ wgt16,
                 const float* __restrict__ gma, const float* __restrict__ bta,
                 float* __restrict__ y)
{
    const int p0 = blockIdx.x * 64;
    const int o0 = blockIdx.y * 128;
    const int b  = blockIdx.z;
    const int tid = threadIdx.x, lane = tid & 31, wid = tid >> 5;
    const int waveM = wid >> 1;          // 0..3 -> 32-row slice (2 subtiles)
    const int waveN = wid & 1;           // 0..1 -> 32-col slice (2 subtiles)
    const bool laneHi = lane >= 16;

    __shared__ __align__(16) _Float16 As[128 * 32];  // [m][k]  8 KB
    __shared__ __align__(16) _Float16 Bs[64 * 32];   // [n][k]  4 KB

    const _Float16* catb = cat16 + (size_t)b * C2HW;

    v8f cf[2][2] = {{{}, {}}, {{}, {}}};

    // A DMA: chunk ci in {tid, tid+256} -> row m = ci/4, 8 halfs at (ci&3)*8
    const int am0 = tid >> 2,        ac0 = (tid & 3) * 8;
    const int am1 = (tid + 256) >> 2, ac1 = ac0;
    // B gather: thread t -> pixel row n = t/4, 8 k at (t&3)*8
    const int bn = tid >> 2, bk = (tid & 3) * 8;
    const int p  = p0 + bn;
    const int ph = p >> 6, pw = p & 63;

    const unsigned ldsA0 = (unsigned)(uintptr_t)(&As[am0 * 32 + ac0]);
    const unsigned ldsA1 = (unsigned)(uintptr_t)(&As[am1 * 32 + ac1]);
    unsigned srcA0 = (unsigned)(((o0 + am0) * KTOT + ac0) * 2);   // byte offsets
    unsigned srcA1 = (unsigned)(((o0 + am1) * KTOT + ac1) * 2);

    for (int k0 = 0; k0 < KTOT; k0 += 32) {
        __syncthreads();
        // async weight DMA into LDS (ASYNCcnt-tracked), 2 x 16 B per lane
        asm volatile("global_load_async_to_lds_b128 %0, %1, %2"
                     :: "v"(ldsA0), "v"(srcA0), "s"(wgt16) : "memory");
        asm volatile("global_load_async_to_lds_b128 %0, %1, %2"
                     :: "v"(ldsA1), "v"(srcA1), "s"(wgt16) : "memory");
        srcA0 += 64;   // advance 32 halfs
        srcA1 += 64;

        // im2col gather (f16) for this block's 64 pixels
#pragma unroll
        for (int j = 0; j < 8; ++j) {
            const int kg = k0 + bk + j;
            const int i  = kg / 9;
            const int r  = kg - i * 9;
            const int kh = r / 3;
            const int kw = r - 3 * kh;
            const int h2 = ph + kh - 1, w2 = pw + kw - 1;
            _Float16 v = (_Float16)0.f;
            if (h2 >= 0 && h2 < 64 && w2 >= 0 && w2 < 64)
                v = catb[(size_t)i * HW + h2 * 64 + w2];
            Bs[bn * 32 + bk + j] = v;
        }
        __builtin_prefetch(&catb[(size_t)((k0 + 32 + bk) / 9) * HW + ph * 64 + pw], 0, 0);

        asm volatile("s_wait_asynccnt 0x0" ::: "memory");
        __syncthreads();

        // A fragments (16x32 f16): lanes<16 -> K 0..7 & 16..23; lanes>=16 -> +8
        const _Float16* ar0 = &As[(32 * waveM + (lane & 15)) * 32 + (laneHi ? 8 : 0)];
        const _Float16* ar1 = ar0 + 16 * 32;
        // B fragments (32x16 f16): col N = lane%16; lanes<16 -> K 0..7 & 16..23; lanes>=16 -> +8
        const _Float16* br0 = &Bs[(32 * waveN + (lane & 15)) * 32 + (laneHi ? 8 : 0)];
        const _Float16* br1 = br0 + 16 * 32;
        v16h af[2], bf[2];
#pragma unroll
        for (int j = 0; j < 8; ++j) {
            af[0][j] = ar0[j];  af[0][j + 8] = ar0[16 + j];
            af[1][j] = ar1[j];  af[1][j + 8] = ar1[16 + j];
        }
#pragma unroll
        for (int j = 0; j < 8; ++j) { bf[0][j] = br0[j]; bf[0][j + 8] = br0[16 + j]; bf[1][j] = br1[j]; bf[1][j + 8] = br1[16 + j]; }

#pragma unroll
        for (int mi = 0; mi < 2; ++mi)
#pragma unroll
            for (int ni = 0; ni < 2; ++ni)
                cf[mi][ni] = __builtin_amdgcn_wmma_f32_16x16x32_f16(
                    false, af[mi], false, bf[ni], (short)0, cf[mi][ni],
                    false, false);
    }

    // D layout: VGPR r -> M = r (lanes 0-15) / 8+r (lanes 16-31); N = lane%16
    const int nc = lane & 15;
    float* yb = y + (size_t)b * OUTC * HW;
#pragma unroll
    for (int mi = 0; mi < 2; ++mi) {
        const int mBase = o0 + 32 * waveM + 16 * mi + (laneHi ? 8 : 0);
#pragma unroll
        for (int r = 0; r < 8; ++r) {
            const int o = mBase + r;
            const float g = gma[o], be = bta[o];
            float v0 = cf[mi][0][r] * g + be; v0 = v0 > 0.f ? v0 : 0.f;
            float v1 = cf[mi][1][r] * g + be; v1 = v1 > 0.f ? v1 : 0.f;
            yb[(size_t)o * HW + p0 + 32 * waveN + nc]      = v0;
            yb[(size_t)o * HW + p0 + 32 * waveN + 16 + nc] = v1;
        }
    }
}

// ---------------------------------------------------------------------------
extern "C" void kernel_launch(void* const* d_in, const int* in_sizes, int n_in,
                              void* d_out, int out_size, void* d_ws, size_t ws_size,
                              hipStream_t stream)
{
    (void)in_sizes; (void)n_in; (void)out_size; (void)ws_size;

    const float* x     = (const float*)d_in[0];
    const float* dctw  = (const float*)d_in[1];
    const float* dmw1  = (const float*)d_in[2];
    const float* dmg   = (const float*)d_in[3];
    const float* dmb   = (const float*)d_in[4];
    const float* dmw2  = (const float*)d_in[5];
    const float* dmb2  = (const float*)d_in[6];
    const float* fusew = (const float*)d_in[7];
    const float* fuseb = (const float*)d_in[8];
    const float* gw1   = (const float*)d_in[9];
    const float* gw2   = (const float*)d_in[10];
    const float* botw  = (const float*)d_in[11];
    const float* botg  = (const float*)d_in[12];
    const float* botb  = (const float*)d_in[13];

    float* yout = (float*)d_out;                      // [8,512,64,64]
    float* fout = yout + (size_t)BSZ * OUTC * HW;     // [8,19,64,64]

    // workspace: cat16 (67.1 MB) | wgt16 (9.4 MB) | local | localF
    _Float16* cat16 = (_Float16*)d_ws;
    _Float16* wgt16 = cat16 + (size_t)BSZ * C2HW;
    float*   dlocal = (float*)(wgt16 + (size_t)OUTC * KTOT);
    float*   dlocF  = dlocal + (size_t)BSZ * NUMG * LKC;

    cvt_wgt_kernel<<<2048, 256, 0, stream>>>(botw, wgt16);
    cvt_x_kernel<<<(BSZ * CHW) / 256, 256, 0, stream>>>(x, cat16);
    stageA_local_kernel<<<dim3(NUMG, BSZ), 256, 0, stream>>>(
        x, dctw, dmw1, dmg, dmb, dmw2, dmb2, dlocal);
    stageA_fuse_kernel<<<dim3(HW / 256, BSZ), 256, 0, stream>>>(
        x, dctw, dmw1, dmg, dmb, dmw2, dmb2, fusew, fuseb, fout);
    gcn_kernel<<<BSZ, 256, 0, stream>>>(dlocal, gw1, gw2, dlocF);
    attn_kernel<<<dim3(NUMG, BSZ), 256, 0, stream>>>(x, dlocF, cat16);
    conv_wmma_kernel<<<dim3(HW / 64, OUTC / 128, BSZ), 256, 0, stream>>>(
        cat16, wgt16, botg, botb, yout);
}
